// MoE_40192303956070
// MI455X (gfx1250) — matrix operand order repaired and checked by reading, not compiled, not run on hardware
//
#include <hip/hip_runtime.h>
#include <hip/hip_bf16.h>
#include <math.h>

// ---------------------------------------------------------------------------
// MoE block for MI455X (gfx1250, wave32, WMMA).
// sl=2048 bs=8 hs=1024 ffn=4096 E=8 top_k=2 capacity=4096
// ---------------------------------------------------------------------------

typedef __attribute__((ext_vector_type(16))) __bf16 v16bf;
typedef __attribute__((ext_vector_type(8)))  float  v8f;

#define HS    1024
#define FFN   4096
#define NEXP  8
#define TOK   16384      // sl*bs
#define CAP   4096       // capacity per expert

union FragU { uint4 q[2]; v16bf v; };

// Single v_perm_b32: pack two fp32 into packed bf16 (truncate).
__device__ __forceinline__ unsigned pack_bf16(float a, float b) {
  return __builtin_amdgcn_perm(__float_as_uint(b), __float_as_uint(a),
                               0x07060302u);
}

__device__ __forceinline__ float gelu_tanh(float x) {
  float x3 = x * x * x;
  return 0.5f * x * (1.0f + tanhf(0.7978845608028654f * (x + 0.044715f * x3)));
}

// ---------------------------------------------------------------------------
// Init: slot_src = 0 (empty slots gather token row 0; never read downstream),
// expert counters = 0.
// ---------------------------------------------------------------------------
__global__ void moe_init(int* __restrict__ slot_src, int* __restrict__ cnt) {
  int i = blockIdx.x * blockDim.x + threadIdx.x;
  if (i < NEXP * CAP) slot_src[i] = 0;
  if (i < NEXP) cnt[i] = 0;
}

// ---------------------------------------------------------------------------
// Router: logits = x @ w_router, softmax, top-2. One wave per token.
// ---------------------------------------------------------------------------
__global__ __launch_bounds__(256) void moe_router(
    const float* __restrict__ X, const float* __restrict__ Wr,
    float* __restrict__ tokW, int* __restrict__ tokE)
{
  __shared__ float wr[NEXP * HS];                 // transposed: [e][h]
  int tid = threadIdx.x;
  for (int j = tid; j < NEXP * HS; j += 256) {
    int e = j & (NEXP - 1);
    int h = j >> 3;
    wr[e * HS + h] = Wr[j];                       // Wr is [h][e] row-major
  }
  __syncthreads();

  int lane = tid & 31;
  int wave = tid >> 5;
  int t = blockIdx.x * 8 + wave;

  float acc[NEXP];
#pragma unroll
  for (int e = 0; e < NEXP; ++e) acc[e] = 0.f;
  const float* xr = X + (size_t)t * HS;
  for (int h = lane; h < HS; h += 32) {
    float xv = xr[h];
#pragma unroll
    for (int e = 0; e < NEXP; ++e) acc[e] += xv * wr[e * HS + h];
  }
#pragma unroll
  for (int off = 16; off > 0; off >>= 1) {
#pragma unroll
    for (int e = 0; e < NEXP; ++e) acc[e] += __shfl_xor(acc[e], off, 32);
  }
  if (lane == 0) {
    float m = acc[0];
#pragma unroll
    for (int e = 1; e < NEXP; ++e) m = fmaxf(m, acc[e]);
    float p[NEXP], s = 0.f;
#pragma unroll
    for (int e = 0; e < NEXP; ++e) { p[e] = __expf(acc[e] - m); s += p[e]; }
    float inv = 1.0f / s;
    int e0 = 0; float b0 = p[0];
#pragma unroll
    for (int e = 1; e < NEXP; ++e) if (p[e] > b0) { b0 = p[e]; e0 = e; }
    int e1 = (e0 == 0) ? 1 : 0; float b1 = p[e1];
#pragma unroll
    for (int e = 0; e < NEXP; ++e)
      if (e != e0 && p[e] > b1) { b1 = p[e]; e1 = e; }
    tokE[t * 2 + 0] = e0; tokE[t * 2 + 1] = e1;
    tokW[t * 2 + 0] = b0 * inv; tokW[t * 2 + 1] = b1 * inv;
  }
}

// ---------------------------------------------------------------------------
// Slot assignment: per (token,k) pair grab a slot in its expert bin.
// ---------------------------------------------------------------------------
__global__ void moe_assign(const int* __restrict__ tokE, int* __restrict__ cnt,
                           int* __restrict__ slot_src, int* __restrict__ tokSlot)
{
  int i = blockIdx.x * blockDim.x + threadIdx.x;   // 0..TOK*2-1
  if (i >= TOK * 2) return;
  int e = tokE[i];
  int pos = atomicAdd(&cnt[e], 1);
  int s = -1;
  if (pos < CAP) { s = e * CAP + pos; slot_src[s] = i >> 1; }
  tokSlot[i] = s;
}

// ---------------------------------------------------------------------------
// Grouped GEMM: C[e*CAP + m, n] = (gelu?) sum_k A_row(m)[k] * Bw[e][k][n]
// Block tile 256x128, 8 waves (4M x 2N), wave tile 64x64 (16 WMMA / K-step).
// K-step 32 (bf16). Global loads for tile k+1 are register-staged during the
// WMMA block of tile k. LDS layouts match ISA 7.12.2 wave32 fragment layouts.
// ---------------------------------------------------------------------------
template<bool GATHER, bool GELU>
__global__ __launch_bounds__(256) void moe_gemm(
    const float* __restrict__ A, const float* __restrict__ Bw,
    float* __restrict__ C, const int* __restrict__ slot_src,
    int N, int K, int nTiles)
{
  __shared__ uint4 ldsA4[256 * 4];   // 256 rows x 16 packed-bf16 dwords (K=32)
  __shared__ uint4 ldsB4[128 * 4];   // 128 cols x 16 packed-bf16 dwords
  unsigned* ldsA = (unsigned*)ldsA4;
  unsigned* ldsB = (unsigned*)ldsB4;

  const int tid   = threadIdx.x;
  const int lane  = tid & 31;
  const int wave  = tid >> 5;
  const int hiL   = lane >> 4;        // lane group 0-15 / 16-31
  const int lo16  = lane & 15;
  const int mWave = wave & 3;         // 4 waves over 256 rows (64 each)
  const int nWave = wave >> 2;        // 2 waves over 128 cols (64 each)

  const int blocksPerExpert = (CAP / 256) * nTiles;
  const int e   = blockIdx.x / blocksPerExpert;
  const int rem = blockIdx.x % blocksPerExpert;
  const int mt  = rem / nTiles;
  const int nt  = rem % nTiles;
  const int mBase = mt * 256;
  const int nBase = nt * 128;
  const float* Be = Bw + (size_t)e * K * N;

  // ---- A loader: 16 dwords/thread; dword i covers row i*16 + tid/16,
  //      K-pair index tid&15 (invariant: 256 % 16 == 0). Unconditional
  //      gather; empty slots read row 0 (result rows never consumed).
  const int aK2 = tid & 15;
  int srcIdx[16];
#pragma unroll
  for (int i = 0; i < 16; ++i) {
    int r = i * 16 + (tid >> 4);
    if (GATHER) {
      int src = slot_src[e * CAP + mBase + r];
      srcIdx[i] = (src < 0) ? 0 : src;
    } else {
      srcIdx[i] = e * CAP + mBase + r;
    }
  }
  // ---- B loader: 8 dwords/thread, coalesced over 128 columns.
  const int nIdx   = tid & 127;
  const int k2base = (tid >> 7) * 8;

  float2 aS[16];
  float2 bS[8];

  auto loadA = [&](int kt) {
#pragma unroll
    for (int i = 0; i < 16; ++i)
      aS[i] = *reinterpret_cast<const float2*>(
          A + (size_t)srcIdx[i] * K + kt + 2 * aK2);
  };
  auto loadB = [&](int kt) {
#pragma unroll
    for (int i = 0; i < 8; ++i) {
      int k = kt + 2 * (k2base + i);
      bS[i].x = Be[(size_t)k * N + nBase + nIdx];
      bS[i].y = Be[(size_t)(k + 1) * N + nBase + nIdx];
    }
  };

  v8f acc[4][4];
#pragma unroll
  for (int mi = 0; mi < 4; ++mi)
#pragma unroll
    for (int ni = 0; ni < 4; ++ni) {
      v8f z = {0.f, 0.f, 0.f, 0.f, 0.f, 0.f, 0.f, 0.f};
      acc[mi][ni] = z;
    }

  loadA(0);
  loadB(0);

  for (int kt = 0; kt < K; kt += 32) {
    // ---- commit staged tile to LDS (fp32 -> packed bf16, v_perm) ----
#pragma unroll
    for (int i = 0; i < 16; ++i) {
      int r = i * 16 + (tid >> 4);
      ldsA[r * 16 + aK2] = pack_bf16(aS[i].x, aS[i].y);
    }
#pragma unroll
    for (int i = 0; i < 8; ++i)
      ldsB[nIdx * 16 + k2base + i] = pack_bf16(bS[i].x, bS[i].y);
    __syncthreads();

    // ---- prefetch next tile into registers (overlaps WMMA below) ----
    if (kt + 32 < K) {
      loadA(kt + 32);
      loadB(kt + 32);
    }

    // ---- fragment loads (two ds_load_b128 each) + 16 WMMAs ----
    const uint4* A4 = (const uint4*)ldsA;
    const uint4* B4 = (const uint4*)ldsB;
    FragU afr[4], bfr[4];
#pragma unroll
    for (int mi = 0; mi < 4; ++mi) {
      int row = mWave * 64 + mi * 16 + lo16;
      afr[mi].q[0] = A4[row * 4 + hiL];        // K 0-7 / 8-15
      afr[mi].q[1] = A4[row * 4 + 2 + hiL];    // K 16-23 / 24-31
    }
#pragma unroll
    for (int ni = 0; ni < 4; ++ni) {
      int col = nWave * 64 + ni * 16 + lo16;
      bfr[ni].q[0] = B4[col * 4 + hiL];        // K 0-7 / 8-15
      bfr[ni].q[1] = B4[col * 4 + 2 + hiL];    // K 16-23 / 24-31
    }
#pragma unroll
    for (int mi = 0; mi < 4; ++mi)
#pragma unroll
      for (int ni = 0; ni < 4; ++ni)
        acc[mi][ni] = __builtin_amdgcn_wmma_f32_16x16x32_bf16(
            false, afr[mi].v, false, bfr[ni].v,
            (short)0, acc[mi][ni], false, false);

    __syncthreads();
  }

  // ---- epilogue: C/D layout VGPR v -> M = v + 8*hiL, N = lo16 ----
#pragma unroll
  for (int mi = 0; mi < 4; ++mi) {
#pragma unroll
    for (int ni = 0; ni < 4; ++ni) {
      int col = nBase + nWave * 64 + ni * 16 + lo16;
#pragma unroll
      for (int v = 0; v < 8; ++v) {
        int r = mBase + mWave * 64 + mi * 16 + v + 8 * hiL;
        float val = acc[mi][ni][v];
        if (GELU) val = gelu_tanh(val);
        C[((size_t)(e * CAP) + r) * N + col] = val;
      }
    }
  }
}

// ---------------------------------------------------------------------------
// Combine: out[t] = w0*y[slot0] + w1*y[slot1] + bias  (deterministic gather)
// ---------------------------------------------------------------------------
__global__ void moe_combine(const float* __restrict__ y,
                            const float* __restrict__ tokW,
                            const int* __restrict__ tokSlot,
                            const float* __restrict__ bias,
                            float* __restrict__ out)
{
  int idx = blockIdx.x * blockDim.x + threadIdx.x;   // float4 index
  int t = (idx * 4) / HS;
  int h = (idx * 4) & (HS - 1);
  int s0 = tokSlot[t * 2], s1 = tokSlot[t * 2 + 1];
  float w0 = tokW[t * 2],  w1 = tokW[t * 2 + 1];
  float4 r = *reinterpret_cast<const float4*>(bias + h);
  if (s0 >= 0) {
    float4 a = *reinterpret_cast<const float4*>(y + (size_t)s0 * HS + h);
    r.x += w0 * a.x; r.y += w0 * a.y; r.z += w0 * a.z; r.w += w0 * a.w;
  }
  if (s1 >= 0) {
    float4 a = *reinterpret_cast<const float4*>(y + (size_t)s1 * HS + h);
    r.x += w1 * a.x; r.y += w1 * a.y; r.z += w1 * a.z; r.w += w1 * a.w;
  }
  *reinterpret_cast<float4*>(out + (size_t)t * HS + h) = r;
}

// ---------------------------------------------------------------------------
extern "C" void kernel_launch(void* const* d_in, const int* in_sizes, int n_in,
                              void* d_out, int out_size, void* d_ws, size_t ws_size,
                              hipStream_t stream)
{
  const float* X    = (const float*)d_in[0];   // [TOK, HS]
  const float* Wr   = (const float*)d_in[1];   // [HS, NEXP]
  const float* W1   = (const float*)d_in[2];   // [NEXP, HS, FFN]
  const float* W2   = (const float*)d_in[3];   // [NEXP, FFN, HS]
  const float* bias = (const float*)d_in[4];   // [HS]
  float* out = (float*)d_out;

  char* ws = (char*)d_ws;
  float* tokW     = (float*)(ws + ((size_t)0 << 17));
  int*   tokE     = (int*)  (ws + ((size_t)1 << 17));
  int*   tokSlot  = (int*)  (ws + ((size_t)2 << 17));
  int*   slot_src = (int*)  (ws + ((size_t)3 << 17));
  int*   cnt      = (int*)  (ws + ((size_t)4 << 17));
  float* hbuf     = (float*)(ws + ((size_t)1 << 20));                    // [NEXP*CAP, FFN] fp32
  float* ybuf     = (float*)(ws + ((size_t)1 << 20)
                                + (size_t)NEXP * CAP * FFN * sizeof(float)); // [NEXP*CAP, HS] fp32

  moe_init  <<<(NEXP * CAP) / 256, 256, 0, stream>>>(slot_src, cnt);
  moe_router<<<TOK / 8,           256, 0, stream>>>(X, Wr, tokW, tokE);
  moe_assign<<<(TOK * 2) / 256,   256, 0, stream>>>(tokE, cnt, slot_src, tokSlot);

  // GEMM1 + gelu: per expert [CAP x FFN] = [CAP x HS] @ [HS x FFN], gathered A
  moe_gemm<true, true><<<NEXP * (CAP / 256) * (FFN / 128), 256, 0, stream>>>(
      X, W1, hbuf, slot_src, FFN, HS, FFN / 128);
  // GEMM2: per expert [CAP x HS] = [CAP x FFN] @ [FFN x HS]
  moe_gemm<false, false><<<NEXP * (CAP / 256) * (HS / 128), 256, 0, stream>>>(
      hbuf, W2, ybuf, nullptr, HS, FFN, HS / 128);

  moe_combine<<<(TOK * HS / 4) / 256, 256, 0, stream>>>(ybuf, tokW, tokSlot, bias, out);
}
